// GNN_64080912056780
// MI455X (gfx1250) — matrix operand order repaired and checked by reading, not compiled, not run on hardware
//
#include <hip/hip_runtime.h>
#include <hip/hip_bf16.h>

// ---------------------------------------------------------------------------
// GATv2 x3 + node/edge heads for MI455X (gfx1250, wave32, WMMA)
// ---------------------------------------------------------------------------

typedef __attribute__((ext_vector_type(16))) __bf16 v16bf;
typedef __attribute__((ext_vector_type(8)))  __bf16 v8bf;
typedef __attribute__((ext_vector_type(8)))  float  v8f;

constexpr int NN = 25000;            // nodes
constexpr int EE = 400000;           // original edges
constexpr int ET = EE + NN;          // edges incl. self loops
constexpr int HD = 8;                // heads
constexpr int CC = 32;               // channels per head (== wave32 width)
constexpr int DD = HD * CC;          // 256
constexpr float NEG_SLOPE = 0.2f;

// ---------------------------------------------------------------- utilities

__global__ __launch_bounds__(256) void k_fill(float* __restrict__ p, long n, float v) {
    long i = (long)blockIdx.x * 256 + threadIdx.x;
    if (i < n) p[i] = v;
}

__device__ inline void atomicMaxF(float* addr, float val) {
    unsigned int* ua = (unsigned int*)addr;
    unsigned int old = *ua;
    while (true) {
        float cur = __uint_as_float(old);
        if (cur >= val) break;
        unsigned int assumed = old;
        old = atomicCAS(ua, assumed, __float_as_uint(val));
        if (old == assumed) break;
    }
}

// ------------------------------------------------ self-loop attr (deg, mean)

__global__ __launch_bounds__(256) void k_deg(const long long* __restrict__ ei,
                                             const float* __restrict__ eattr,
                                             float* __restrict__ deg,
                                             float* __restrict__ lsum) {
    int e = blockIdx.x * 256 + threadIdx.x;
    if (e >= EE) return;
    int dst = (int)ei[EE + e];
    atomicAdd(&deg[dst], 1.0f);
    atomicAdd(&lsum[dst], eattr[e]);
}

__global__ __launch_bounds__(256) void k_loopattr(const float* __restrict__ lsum,
                                                  const float* __restrict__ deg,
                                                  float* __restrict__ loop_attr) {
    int n = blockIdx.x * 256 + threadIdx.x;
    if (n >= NN) return;
    loop_attr[n] = lsum[n] / fmaxf(deg[n], 1.0f);
}

// ------------------------------------------------------------- bf16 staging

// x (N,3) -> x16 (N,32) zero-padded
__global__ __launch_bounds__(256) void k_prep_x0(const float* __restrict__ x,
                                                 __bf16* __restrict__ x16) {
    long i = (long)blockIdx.x * 256 + threadIdx.x;
    if (i >= (long)NN * 32) return;
    int n = (int)(i >> 5), k = (int)(i & 31);
    x16[i] = (k < 3) ? (__bf16)x[n * 3 + k] : (__bf16)0.0f;
}

// W (K,256) row-major -> Wt (256,Kp) bf16, zero-padded in K
__global__ __launch_bounds__(256) void k_prep_wt(const float* __restrict__ W,
                                                 __bf16* __restrict__ Wt,
                                                 int K, int Kp) {
    int i = blockIdx.x * 256 + threadIdx.x;
    if (i >= 256 * Kp) return;
    int c = i / Kp, k = i % Kp;
    Wt[i] = (k < K) ? (__bf16)W[k * 256 + c] : (__bf16)0.0f;
}

// -------------------------------------------------------------- WMMA GEMM
// Y[rows,256] = X[rows,K](bf16) @ W[K,256] + bias;  Wt is W column-major bf16.
// Block = 256 threads = 8 waves; block tile = 32 rows x 256 cols;
// wave tile = 16 rows x 64 cols (4 accumulators), K step 32.

template <int K>
__global__ __launch_bounds__(256) void k_gemm(const __bf16* __restrict__ X,
                                              const __bf16* __restrict__ Wt,
                                              const float* __restrict__ bias,
                                              float* __restrict__ Y, int rows) {
    const int lane    = threadIdx.x & 31;
    const int wave    = threadIdx.x >> 5;
    const int m0      = blockIdx.x * 32 + (wave >> 2) * 16;
    const int colbase = (wave & 3) * 64;
    const int mrow    = m0 + (lane & 15);
    const int khalfA  = (lane >> 4) * 8;    // A K-half per ISA 16-bit A layout
    const int khalfB  = (lane >> 4) * 8;    // B K-half, same k order as A
    const int bcol    = colbase + (lane & 15);
    const bool fullTile = ((blockIdx.x + 1) * 32 <= rows);
    const bool mvalid   = (mrow < rows);

    v8f acc[4] = {};

    for (int k0 = 0; k0 < K; k0 += 32) {
        v16bf a;
        if (mvalid) {
            const v8bf alo = *(const v8bf*)(X + (size_t)mrow * K + k0 + khalfA);
            const v8bf ahi = *(const v8bf*)(X + (size_t)mrow * K + k0 + khalfA + 16);
#pragma unroll
            for (int i = 0; i < 8; ++i) { a[i] = alo[i]; a[i + 8] = ahi[i]; }
        } else {
#pragma unroll
            for (int i = 0; i < 16; ++i) a[i] = (__bf16)0.0f;
        }
#pragma unroll
        for (int j = 0; j < 4; ++j) {
            const __bf16* bp = Wt + (size_t)(bcol + j * 16) * K + k0 + khalfB;
            const v8bf blo = *(const v8bf*)(bp);
            const v8bf bhi = *(const v8bf*)(bp + 16);
            v16bf b;
#pragma unroll
            for (int i = 0; i < 8; ++i) { b[i] = blo[i]; b[i + 8] = bhi[i]; }
            acc[j] = __builtin_amdgcn_wmma_f32_16x16x32_bf16(
                false, a, false, b, (short)0, acc[j], false, false);
        }
    }

    const int r0 = m0 + ((lane >> 4) << 3);   // C/D: lanes 16-31 hold M=8..15
    if (fullTile) {
        // branch-free epilogue: bias hoisted, unconditional coalesced stores
#pragma unroll
        for (int j = 0; j < 4; ++j) {
            const int col = colbase + j * 16 + (lane & 15);
            const float bcv = bias[col];
            float* yp = Y + (size_t)r0 * 256 + col;
#pragma unroll
            for (int r = 0; r < 8; ++r) yp[(size_t)r * 256] = acc[j][r] + bcv;
        }
    } else {
#pragma unroll
        for (int j = 0; j < 4; ++j) {
            const int col = colbase + j * 16 + (lane & 15);
            const float bcv = bias[col];
#pragma unroll
            for (int r = 0; r < 8; ++r) {
                const int row = r0 + r;
                if (row < rows) Y[(size_t)row * 256 + col] = acc[j][r] + bcv;
            }
        }
    }
}

// ------------------------------------------------------ attention, 3 passes
// One wave per edge; lane == channel (CC == 32 == wave width).

__global__ __launch_bounds__(256) void k_edge_logit(const long long* __restrict__ ei,
                                                    const float* __restrict__ eattr,
                                                    const float* __restrict__ loop_attr,
                                                    const float* __restrict__ xl,
                                                    const float* __restrict__ xr,
                                                    const float* __restrict__ We,
                                                    const float* __restrict__ att,
                                                    float* __restrict__ logit,
                                                    float* __restrict__ lmax) {
    const int e = blockIdx.x * 8 + (threadIdx.x >> 5);
    if (e >= ET) return;
    const int lane = threadIdx.x & 31;
    int src, dst; float ea;
    if (e < EE) { src = (int)ei[e]; dst = (int)ei[EE + e]; ea = eattr[e]; }
    else        { src = dst = e - EE; ea = loop_attr[e - EE]; }
    const float* pl = xl + (size_t)src * DD;
    const float* pr = xr + (size_t)dst * DD;
#pragma unroll
    for (int h = 0; h < HD; ++h) {
        const int idx = h * CC + lane;
        float m = pl[idx] + pr[idx] + ea * We[idx];
        m = m > 0.0f ? m : NEG_SLOPE * m;
        float v = m * att[idx];
#pragma unroll
        for (int o = 16; o > 0; o >>= 1) v += __shfl_xor(v, o, 32);
        if (lane == 0) {
            logit[(size_t)e * HD + h] = v;
            atomicMaxF(&lmax[(size_t)dst * HD + h], v);
        }
    }
}

__global__ __launch_bounds__(256) void k_edge_exp(const long long* __restrict__ ei,
                                                  float* __restrict__ logit,
                                                  const float* __restrict__ lmax,
                                                  float* __restrict__ denom) {
    long t = (long)blockIdx.x * 256 + threadIdx.x;
    if (t >= (long)ET * HD) return;
    int e = (int)(t >> 3), h = (int)(t & 7);
    int dst = (e < EE) ? (int)ei[EE + e] : (e - EE);
    float a = __expf(logit[t] - lmax[(size_t)dst * HD + h]);
    logit[t] = a;                       // reuse buffer as unnormalized alpha
    atomicAdd(&denom[(size_t)dst * HD + h], a);
}

__global__ __launch_bounds__(256) void k_edge_scatter(const long long* __restrict__ ei,
                                                      const float* __restrict__ abuf,
                                                      const float* __restrict__ denom,
                                                      const float* __restrict__ xl,
                                                      float* __restrict__ out) {
    const int e = blockIdx.x * 8 + (threadIdx.x >> 5);
    if (e >= ET) return;
    const int lane = threadIdx.x & 31;
    int src, dst;
    if (e < EE) { src = (int)ei[e]; dst = (int)ei[EE + e]; }
    else        { src = dst = e - EE; }
    const float* pl = xl + (size_t)src * DD;
    float* po = out + (size_t)dst * DD;
#pragma unroll
    for (int h = 0; h < HD; ++h) {
        float alpha = abuf[(size_t)e * HD + h] / denom[(size_t)dst * HD + h];
        atomicAdd(&po[h * CC + lane], pl[h * CC + lane] * alpha);
    }
}

// -------------------------------------------------------------- layer post

// concat layers: next input x16 = bf16(relu(out + bias))
__global__ __launch_bounds__(256) void k_post_concat(const float* __restrict__ out,
                                                     const float* __restrict__ bias,
                                                     __bf16* __restrict__ x16) {
    long i = (long)blockIdx.x * 256 + threadIdx.x;
    if (i >= (long)NN * DD) return;
    float v = fmaxf(out[i] + bias[(int)(i & 255)], 0.0f);
    x16[i] = (__bf16)v;
}

// final layer: hfin = relu(mean_h(out) + bias)
__global__ __launch_bounds__(256) void k_post_mean(const float* __restrict__ out,
                                                   const float* __restrict__ bias,
                                                   float* __restrict__ hfin) {
    long i = (long)blockIdx.x * 256 + threadIdx.x;
    if (i >= (long)NN * CC) return;
    int n = (int)(i >> 5), c = (int)(i & 31);
    float s = 0.0f;
#pragma unroll
    for (int h = 0; h < HD; ++h) s += out[(size_t)n * DD + h * CC + c];
    hfin[i] = fmaxf(s * 0.125f + bias[c], 0.0f);
}

// ------------------------------------------------------------------- heads

__device__ inline void mlp2_logsm(const float* __restrict__ f,
                                  const float* __restrict__ W1, const float* __restrict__ b1,
                                  const float* __restrict__ W2, const float* __restrict__ b2,
                                  float* __restrict__ o) {
    float t[CC];
#pragma unroll 4
    for (int j = 0; j < CC; ++j) {
        float s = b1[j];
        for (int c = 0; c < CC; ++c) s += f[c] * W1[c * CC + j];
        t[j] = fmaxf(s, 0.0f);
    }
    float l0 = b2[0], l1 = b2[1];
    for (int j = 0; j < CC; ++j) { l0 += t[j] * W2[j * 2 + 0]; l1 += t[j] * W2[j * 2 + 1]; }
    float mx  = fmaxf(l0, l1);
    float lse = mx + __logf(__expf(l0 - mx) + __expf(l1 - mx));
    o[0] = l0 - lse;
    o[1] = l1 - lse;
}

__global__ __launch_bounds__(256) void k_node_head(const float* __restrict__ hfin,
                                                   const float* __restrict__ W1,
                                                   const float* __restrict__ b1,
                                                   const float* __restrict__ W2,
                                                   const float* __restrict__ b2,
                                                   float* __restrict__ outp) {
    int n = blockIdx.x * 256 + threadIdx.x;
    if (n >= NN) return;
    mlp2_logsm(hfin + (size_t)n * CC, W1, b1, W2, b2, outp + (size_t)n * 2);
}

__global__ __launch_bounds__(256) void k_edge_head(const float* __restrict__ hfin,
                                                   const long long* __restrict__ ei,
                                                   const float* __restrict__ W1,
                                                   const float* __restrict__ b1,
                                                   const float* __restrict__ W2,
                                                   const float* __restrict__ b2,
                                                   float* __restrict__ outp) {
    int e = blockIdx.x * 256 + threadIdx.x;
    if (e >= EE) return;
    int s = (int)ei[e], d = (int)ei[EE + e];
    float f[CC];
#pragma unroll
    for (int c = 0; c < CC; ++c) f[c] = hfin[(size_t)s * CC + c] + hfin[(size_t)d * CC + c];
    mlp2_logsm(f, W1, b1, W2, b2, outp + (size_t)e * 2);
}

// ------------------------------------------------------------------ launch

extern "C" void kernel_launch(void* const* d_in, const int* in_sizes, int n_in,
                              void* d_out, int out_size, void* d_ws, size_t ws_size,
                              hipStream_t stream) {
    (void)in_sizes; (void)n_in; (void)out_size; (void)ws_size;

    const float*     x     = (const float*)d_in[0];
    const long long* ei    = (const long long*)d_in[1];
    const float*     eattr = (const float*)d_in[2];
    const float *Wl[3], *bl[3], *Wr[3], *br[3], *We[3], *att[3], *bias[3];
    for (int li = 0; li < 3; ++li) {
        int b = 3 + li * 7;
        Wl[li]   = (const float*)d_in[b + 0];
        bl[li]   = (const float*)d_in[b + 1];
        Wr[li]   = (const float*)d_in[b + 2];
        br[li]   = (const float*)d_in[b + 3];
        We[li]   = (const float*)d_in[b + 4];
        att[li]  = (const float*)d_in[b + 5];
        bias[li] = (const float*)d_in[b + 6];
    }
    const float* nh_W1 = (const float*)d_in[24];
    const float* nh_b1 = (const float*)d_in[25];
    const float* nh_W2 = (const float*)d_in[26];
    const float* nh_b2 = (const float*)d_in[27];
    const float* eh_W1 = (const float*)d_in[28];
    const float* eh_b1 = (const float*)d_in[29];
    const float* eh_W2 = (const float*)d_in[30];
    const float* eh_b2 = (const float*)d_in[31];

    // ---- workspace carving
    char* base = (char*)d_ws;
    size_t off = 0;
    auto carve = [&](size_t bytes) -> void* {
        off = (off + 255) & ~(size_t)255;
        void* p = base + off;
        off += bytes;
        return p;
    };
    float*  xl        = (float*)carve((size_t)NN * DD * 4);
    float*  xr        = (float*)carve((size_t)NN * DD * 4);
    float*  outb      = (float*)carve((size_t)NN * DD * 4);
    float*  logit     = (float*)carve((size_t)ET * HD * 4);
    float*  lmax      = (float*)carve((size_t)NN * HD * 4);
    float*  denom     = (float*)carve((size_t)NN * HD * 4);
    float*  deg       = (float*)carve((size_t)NN * 4);
    float*  lsum      = (float*)carve((size_t)NN * 4);
    float*  loop_attr = (float*)carve((size_t)NN * 4);
    float*  hfin      = (float*)carve((size_t)NN * CC * 4);
    __bf16* x16       = (__bf16*)carve((size_t)NN * DD * 2);
    __bf16* wtl       = (__bf16*)carve((size_t)256 * 256 * 2);
    __bf16* wtr       = (__bf16*)carve((size_t)256 * 256 * 2);

    auto blocks = [](long n) { return (unsigned)((n + 255) / 256); };

    // ---- self-loop edge attr (mean of incoming attrs)
    k_fill<<<blocks(NN), 256, 0, stream>>>(deg, NN, 0.0f);
    k_fill<<<blocks(NN), 256, 0, stream>>>(lsum, NN, 0.0f);
    k_deg<<<blocks(EE), 256, 0, stream>>>(ei, eattr, deg, lsum);
    k_loopattr<<<blocks(NN), 256, 0, stream>>>(lsum, deg, loop_attr);

    // ---- layer-0 input staging (K padded 3 -> 32)
    k_prep_x0<<<blocks((long)NN * 32), 256, 0, stream>>>(x, x16);

    const unsigned gemm_grid = (NN + 31) / 32;
    const unsigned edge_grid = (ET + 7) / 8;

    for (int li = 0; li < 3; ++li) {
        const int K  = (li == 0) ? 3 : 256;
        const int Kp = (li == 0) ? 32 : 256;

        k_prep_wt<<<blocks(256 * Kp), 256, 0, stream>>>(Wl[li], wtl, K, Kp);
        k_prep_wt<<<blocks(256 * Kp), 256, 0, stream>>>(Wr[li], wtr, K, Kp);

        if (li == 0) {
            k_gemm<32><<<gemm_grid, 256, 0, stream>>>(x16, wtl, bl[li], xl, NN);
            k_gemm<32><<<gemm_grid, 256, 0, stream>>>(x16, wtr, br[li], xr, NN);
        } else {
            k_gemm<256><<<gemm_grid, 256, 0, stream>>>(x16, wtl, bl[li], xl, NN);
            k_gemm<256><<<gemm_grid, 256, 0, stream>>>(x16, wtr, br[li], xr, NN);
        }

        k_fill<<<blocks((long)NN * HD), 256, 0, stream>>>(lmax, (long)NN * HD, -3.0e38f);
        k_fill<<<blocks((long)NN * HD), 256, 0, stream>>>(denom, (long)NN * HD, 0.0f);
        k_fill<<<blocks((long)NN * DD), 256, 0, stream>>>(outb, (long)NN * DD, 0.0f);

        k_edge_logit<<<edge_grid, 256, 0, stream>>>(ei, eattr, loop_attr, xl, xr,
                                                    We[li], att[li], logit, lmax);
        k_edge_exp<<<blocks((long)ET * HD), 256, 0, stream>>>(ei, logit, lmax, denom);
        k_edge_scatter<<<edge_grid, 256, 0, stream>>>(ei, logit, denom, xl, outb);

        if (li < 2)
            k_post_concat<<<blocks((long)NN * DD), 256, 0, stream>>>(outb, bias[li], x16);
        else
            k_post_mean<<<blocks((long)NN * CC), 256, 0, stream>>>(outb, bias[li], hfin);
    }

    float* node_out = (float*)d_out;            // (N,2)
    float* edge_out = node_out + (size_t)NN * 2; // (E,2)
    k_node_head<<<blocks(NN), 256, 0, stream>>>(hfin, nh_W1, nh_b1, nh_W2, nh_b2, node_out);
    k_edge_head<<<blocks(EE), 256, 0, stream>>>(hfin, ei, eh_W1, eh_b1, eh_W2, eh_b2, edge_out);
}
